// GATWithJK_61229053772419
// MI455X (gfx1250) — matrix operand order repaired and checked by reading, not compiled, not run on hardware
//
#include <hip/hip_runtime.h>
#include <stdint.h>

typedef __attribute__((ext_vector_type(16))) __bf16 v16bf;
typedef __attribute__((ext_vector_type(8)))  float  v8f;

static constexpr int   NN    = 50000;
static constexpr int   EE    = 800000;
static constexpr int   ETOT  = EE + NN;       // with self loops
static constexpr int   FIN   = 128;
static constexpr int   HC    = 256;
static constexpr int   MT    = NN / 16;       // 3125 row tiles (exact)
static constexpr float SLOPE = 0.2f;

// ---------------------------------------------------------------- helpers
__device__ __forceinline__ unsigned short f2bf(float f) {
  unsigned u = __float_as_uint(f);
  u += 0x7FFFu + ((u >> 16) & 1u);            // round-to-nearest-even
  return (unsigned short)(u >> 16);
}
__device__ __forceinline__ float sigm(float x) { return 1.f / (1.f + __expf(-x)); }
__device__ __forceinline__ void atomicMaxF(float* addr, float val) {
  if (val >= 0.f) atomicMax((int*)addr, __float_as_int(val));
  else            atomicMin((unsigned int*)addr, __float_as_uint(val));
}

// ---------------------------------------------------- bf16 fragment packing
// A fragment (16x32, 16-bit): lane l -> row = l&15; half = l>>4;
//   elems 0..15 -> K = 16*half + e
__global__ __launch_bounds__(256) void pack_a_concat(
    const float* __restrict__ A0, int K0,
    const float* __restrict__ A1, int K1, int bcast1,
    unsigned short* __restrict__ PA, int M, int Ktiles)
{
  int tile = blockIdx.x * 8 + (threadIdx.x >> 5);
  int lane = threadIdx.x & 31;
  int total = ((M + 15) >> 4) * Ktiles;
  if (tile >= total) return;
  int mt = tile / Ktiles, kt = tile % Ktiles;
  int m = mt * 16 + (lane & 15);
  int half = lane >> 4;
  union { unsigned short s[16]; uint4 q[2]; } u;
#pragma unroll
  for (int e = 0; e < 16; ++e) {
    int k = kt * 32 + 16 * half + e;
    float f = 0.f;
    if (m < M) {
      if (k < K0)      f = A0[(size_t)m * K0 + k];
      else if (bcast1) f = A1[k - K0];
      else             f = A1[(size_t)m * K1 + (k - K0)];
    }
    u.s[e] = f2bf(f);
  }
  uint4* dst = (uint4*)(PA + ((size_t)tile * 32 + lane) * 16);
  dst[0] = u.q[0]; dst[1] = u.q[1];
}

// B fragment (32x16, 16-bit): lane l -> col = l&15; half = l>>4; K = 16*half + e
__global__ __launch_bounds__(256) void pack_b(
    const float* __restrict__ W, unsigned short* __restrict__ PB, int K, int Nc)
{
  int Ntiles = Nc >> 4, Ktiles = K >> 5;
  int tile = blockIdx.x * 8 + (threadIdx.x >> 5);
  if (tile >= Ktiles * Ntiles) return;
  int kt = tile / Ntiles, nt = tile % Ntiles;
  int lane = threadIdx.x & 31;
  int n = nt * 16 + (lane & 15);
  int half = lane >> 4;
  union { unsigned short s[16]; uint4 q[2]; } u;
#pragma unroll
  for (int e = 0; e < 16; ++e) {
    int k = kt * 32 + 16 * half + e;
    u.s[e] = f2bf(W[(size_t)k * Nc + n]);
  }
  uint4* dst = (uint4*)(PB + ((size_t)tile * 32 + lane) * 16);
  dst[0] = u.q[0]; dst[1] = u.q[1];
}

// B = [Wih.T ; Whh.T] : [512,1024], Wih/Whh stored [1024,256] row-major
__global__ __launch_bounds__(256) void pack_b_lstm(
    const float* __restrict__ Wih, const float* __restrict__ Whh,
    unsigned short* __restrict__ PB)
{
  const int Ntiles = 64;
  int tile = blockIdx.x * 8 + (threadIdx.x >> 5);
  if (tile >= 16 * Ntiles) return;
  int kt = tile / Ntiles, nt = tile % Ntiles;
  int lane = threadIdx.x & 31;
  int n = nt * 16 + (lane & 15);
  int half = lane >> 4;
  union { unsigned short s[16]; uint4 q[2]; } u;
#pragma unroll
  for (int e = 0; e < 16; ++e) {
    int k = kt * 32 + 16 * half + e;
    float f = (k < HC) ? Wih[(size_t)n * HC + k] : Whh[(size_t)n * HC + (k - HC)];
    u.s[e] = f2bf(f);
  }
  uint4* dst = (uint4*)(PB + ((size_t)tile * 32 + lane) * 16);
  dst[0] = u.q[0]; dst[1] = u.q[1];
}

// ------------------------------------------------------------- WMMA GEMM
// 4 waves/block; wave -> 2 row tiles x 4 N tiles.
// Ping-pong double-buffered k-loop (Ktiles must be even): no register
// rotation copies, so no WMMA->VALU hazard NOPs in the steady state.
__global__ __launch_bounds__(128) void wmma_gemm_bf16(
    const unsigned short* __restrict__ PA, const unsigned short* __restrict__ PB,
    float* __restrict__ C, const float* __restrict__ bias,
    int Mtiles, int Ktiles, int Ntiles, int Nc, int relu)
{
  int lane = threadIdx.x & 31;
  int pairIdx = blockIdx.x * 4 + (threadIdx.x >> 5);
  int mt0 = pairIdx * 2;
  if (mt0 >= Mtiles) return;
  int mt1 = mt0 + 1;
  bool has1 = (mt1 < Mtiles);
  int mt1c = has1 ? mt1 : mt0;
  int ntb = blockIdx.y * 4;
  const v16bf* pa = (const v16bf*)PA;
  const v16bf* pb = (const v16bf*)PB;
  const v16bf* pa0 = pa + (size_t)mt0 * Ktiles * 32 + lane;
  const v16bf* pa1 = pa + (size_t)mt1c * Ktiles * 32 + lane;
  v8f acc0[4] = {}, acc1[4] = {};
  v16bf a0A, a1A, bA[4];                     // buffer set A
  v16bf a0B, a1B, bB[4];                     // buffer set B

  auto loadset = [&](int kt, v16bf& A0, v16bf& A1, v16bf (&B)[4]) {
    A0 = pa0[(size_t)kt * 32];
    A1 = pa1[(size_t)kt * 32];
    const v16bf* pbk = pb + ((size_t)kt * Ntiles + ntb) * 32 + lane;
#pragma unroll
    for (int j = 0; j < 4; ++j) B[j] = pbk[(size_t)j * 32];
  };
  auto mmaset = [&](const v16bf& A0, const v16bf& A1, const v16bf (&B)[4]) {
#pragma unroll
    for (int j = 0; j < 4; ++j) {
      acc0[j] = __builtin_amdgcn_wmma_f32_16x16x32_bf16(
          false, A0, false, B[j], (short)0, acc0[j], false, false);
      acc1[j] = __builtin_amdgcn_wmma_f32_16x16x32_bf16(
          false, A1, false, B[j], (short)0, acc1[j], false, false);
    }
  };

  loadset(0, a0A, a1A, bA);
  for (int kt = 0; kt < Ktiles; kt += 2) {
    loadset(kt + 1, a0B, a1B, bB);
    mmaset(a0A, a1A, bA);
    loadset((kt + 2 < Ktiles) ? kt + 2 : 0, a0A, a1A, bA);
    mmaset(a0B, a1B, bB);
  }

  int half = lane >> 4, col = lane & 15;
#pragma unroll
  for (int j = 0; j < 4; ++j) {
    int ncol = (ntb + j) * 16 + col;
    float bv = bias ? bias[ncol] : 0.f;
#pragma unroll
    for (int r = 0; r < 8; ++r) {
      float v = acc0[j][r] + bv;
      if (relu) v = fmaxf(v, 0.f);
      C[(size_t)(mt0 * 16 + 8 * half + r) * Nc + ncol] = v;
    }
    if (has1) {
#pragma unroll
      for (int r = 0; r < 8; ++r) {
        float v = acc1[j][r] + bv;
        if (relu) v = fmaxf(v, 0.f);
        C[(size_t)(mt1 * 16 + 8 * half + r) * Nc + ncol] = v;
      }
    }
  }
}

// ------------------------------------------------ fused LSTM step epilogue
__device__ __forceinline__ void lstm_epi(const v8f (&acc)[4], int mt, int half,
                                         int ch, float* __restrict__ hstate,
                                         float* __restrict__ cstate,
                                         const float* __restrict__ bias)
{
#pragma unroll
  for (int r = 0; r < 8; ++r) {
    int row = mt * 16 + 8 * half + r;
    size_t idx = (size_t)row * HC + ch;
    float gi = acc[0][r] + bias[ch];
    float gf = acc[1][r] + bias[256 + ch];
    float gg = acc[2][r] + bias[512 + ch];
    float go = acc[3][r] + bias[768 + ch];
    float cn = sigm(gf) * cstate[idx] + sigm(gi) * tanhf(gg);
    cstate[idx] = cn;
    hstate[idx] = sigm(go) * tanhf(cn);
  }
}

// Fused LSTM step: g = [x,h]@[Wih;Whh].T + bias, gates applied in epilogue.
// Wave: 2 row tiles x gate tiles {nt, nt+16, nt+32, nt+48}; ping-pong k-loop.
__global__ __launch_bounds__(128) void wmma_lstm_step(
    const unsigned short* __restrict__ PA, const unsigned short* __restrict__ PB,
    float* __restrict__ hstate, float* __restrict__ cstate,
    const float* __restrict__ bias, int Mtiles)
{
  const int Ktiles = 16, Ntiles = 64;
  int lane = threadIdx.x & 31;
  int pairIdx = blockIdx.x * 4 + (threadIdx.x >> 5);
  int mt0 = pairIdx * 2;
  if (mt0 >= Mtiles) return;
  int mt1 = mt0 + 1;
  bool has1 = (mt1 < Mtiles);
  int mt1c = has1 ? mt1 : mt0;
  int nt = blockIdx.y;                       // 0..15 channel tile
  const v16bf* pa = (const v16bf*)PA;
  const v16bf* pb = (const v16bf*)PB;
  const v16bf* pa0 = pa + (size_t)mt0 * Ktiles * 32 + lane;
  const v16bf* pa1 = pa + (size_t)mt1c * Ktiles * 32 + lane;
  v8f acc0[4] = {}, acc1[4] = {};            // i, f, g, o
  v16bf a0A, a1A, bA[4];
  v16bf a0B, a1B, bB[4];

  auto loadset = [&](int kt, v16bf& A0, v16bf& A1, v16bf (&B)[4]) {
    A0 = pa0[(size_t)kt * 32];
    A1 = pa1[(size_t)kt * 32];
#pragma unroll
    for (int j = 0; j < 4; ++j)
      B[j] = pb[((size_t)kt * Ntiles + nt + 16 * j) * 32 + lane];
  };
  auto mmaset = [&](const v16bf& A0, const v16bf& A1, const v16bf (&B)[4]) {
#pragma unroll
    for (int j = 0; j < 4; ++j) {
      acc0[j] = __builtin_amdgcn_wmma_f32_16x16x32_bf16(
          false, A0, false, B[j], (short)0, acc0[j], false, false);
      acc1[j] = __builtin_amdgcn_wmma_f32_16x16x32_bf16(
          false, A1, false, B[j], (short)0, acc1[j], false, false);
    }
  };

  loadset(0, a0A, a1A, bA);
  for (int kt = 0; kt < Ktiles; kt += 2) {
    loadset(kt + 1, a0B, a1B, bB);
    mmaset(a0A, a1A, bA);
    loadset((kt + 2 < Ktiles) ? kt + 2 : 0, a0A, a1A, bA);
    mmaset(a0B, a1B, bB);
  }

  int half = lane >> 4, col = lane & 15;
  int ch = nt * 16 + col;
  lstm_epi(acc0, mt0, half, ch, hstate, cstate, bias);
  if (has1) lstm_epi(acc1, mt1, half, ch, hstate, cstate, bias);
}

// ------------------------------------------------------------ GAT pieces
__global__ __launch_bounds__(256) void node_attn_k(const float* __restrict__ h,
    const float* __restrict__ asr, const float* __restrict__ adt,
    float* __restrict__ as_, float* __restrict__ ad_, int n)
{
  int node = blockIdx.x * 8 + (threadIdx.x >> 5);
  int lane = threadIdx.x & 31;
  if (node >= n) return;
  const float* hp = h + (size_t)node * HC;
  float ss = 0.f, sd = 0.f;
#pragma unroll
  for (int i = 0; i < 8; ++i) {
    float hv = hp[lane * 8 + i];
    ss += hv * asr[lane * 8 + i];
    sd += hv * adt[lane * 8 + i];
  }
  for (int off = 1; off < 8; off <<= 1) {
    ss += __shfl_xor(ss, off, 32);
    sd += __shfl_xor(sd, off, 32);
  }
  if ((lane & 7) == 0) {
    int head = lane >> 3;
    as_[node * 4 + head] = ss;
    ad_[node * 4 + head] = sd;
  }
}

__device__ __forceinline__ void edge_sd(const int* __restrict__ ei, int e,
                                        int& s, int& d) {
  if (e < EE) { s = ei[e]; d = ei[EE + e]; } else { s = d = e - EE; }
}

__global__ __launch_bounds__(256) void edge_max_k(const int* __restrict__ ei,
    const float* __restrict__ as_, const float* __restrict__ ad_,
    float* __restrict__ m)
{
  int tid = blockIdx.x * 256 + threadIdx.x;
  if (tid >= ETOT * 4) return;
  int e = tid >> 2, hh = tid & 3, s, d;
  edge_sd(ei, e, s, d);
  float v = as_[s * 4 + hh] + ad_[d * 4 + hh];
  v = (v > 0.f) ? v : SLOPE * v;
  atomicMaxF(m + d * 4 + hh, v);
}

__global__ __launch_bounds__(256) void edge_exp_k(const int* __restrict__ ei,
    const float* __restrict__ as_, const float* __restrict__ ad_,
    const float* __restrict__ m, float* __restrict__ z, float* __restrict__ p)
{
  int tid = blockIdx.x * 256 + threadIdx.x;
  if (tid >= ETOT * 4) return;
  int e = tid >> 2, hh = tid & 3, s, d;
  edge_sd(ei, e, s, d);
  float v = as_[s * 4 + hh] + ad_[d * 4 + hh];
  v = (v > 0.f) ? v : SLOPE * v;
  float pe = __expf(v - m[d * 4 + hh]);
  p[tid] = pe;
  atomicAdd(z + d * 4 + hh, pe);
}

__global__ __launch_bounds__(256) void edge_aggr_k(const int* __restrict__ ei,
    const float* __restrict__ h, const float* __restrict__ p,
    const float* __restrict__ z, float* __restrict__ out)
{
  int e = blockIdx.x;
  int t = threadIdx.x, hh = t >> 6, s, d;
  edge_sd(ei, e, s, d);
  float alpha = p[(size_t)e * 4 + hh] / z[d * 4 + hh];
  atomicAdd(out + (size_t)d * HC + t, h[(size_t)s * HC + t] * alpha);
}

// ------------------------------------------------------------ misc kernels
__global__ void fill_f32_k(float* p, float v, size_t n) {
  size_t i = (size_t)blockIdx.x * 256 + threadIdx.x;
  if (i < n) p[i] = v;
}
__global__ void fill_ptr_k(float* p, const float* s, size_t n) {
  size_t i = (size_t)blockIdx.x * 256 + threadIdx.x;
  if (i < n) p[i] = s[0];
}
__global__ void add_vec_k(const float* a, const float* b, float* o, int n) {
  int i = blockIdx.x * 256 + threadIdx.x;
  if (i < n) o[i] = a[i] + b[i];
}
__global__ void bias_rows_k(float* out, const float* __restrict__ b, size_t total) {
  size_t i = (size_t)blockIdx.x * 256 + threadIdx.x;
  if (i < total) out[i] = b[i & 255];
}
__global__ void relu_k(float* p, size_t n) {
  size_t i = (size_t)blockIdx.x * 256 + threadIdx.x;
  if (i < n) p[i] = fmaxf(p[i], 0.f);
}

__global__ __launch_bounds__(256) void score_accum_k(const float* __restrict__ h,
    const float* __restrict__ aw, float* __restrict__ sc, int n)
{
  int node = blockIdx.x * 8 + (threadIdx.x >> 5);
  int lane = threadIdx.x & 31;
  if (node >= n) return;
  const float* hp = h + (size_t)node * HC;
  float s = 0.f;
#pragma unroll
  for (int i = 0; i < 8; ++i) s += hp[lane * 8 + i] * aw[lane * 8 + i];
  for (int off = 16; off > 0; off >>= 1) s += __shfl_xor(s, off, 32);
  if (lane == 0) sc[node] += s;
}

__global__ void jk_softmax_k(const float* __restrict__ x1, const float* __restrict__ x2,
                             const float* __restrict__ x3, const float* __restrict__ sc,
                             float* __restrict__ xj, int n)
{
  size_t tid = (size_t)blockIdx.x * 256 + threadIdx.x;
  if (tid >= (size_t)n * HC) return;
  int node = (int)(tid >> 8);
  float s0 = sc[node], s1 = sc[n + node], s2 = sc[2 * n + node];
  float mx = fmaxf(s0, fmaxf(s1, s2));
  float e0 = __expf(s0 - mx), e1 = __expf(s1 - mx), e2 = __expf(s2 - mx);
  float inv = 1.f / (e0 + e1 + e2);
  xj[tid] = (e0 * x1[tid] + e1 * x2[tid] + e2 * x3[tid]) * inv;
}

__global__ void load_pool_k(const float* __restrict__ xj, const int* __restrict__ li,
                            float* __restrict__ lv, int nloads)
{
  int ch = threadIdx.x;
  float s = 0.f;
  for (int i = 0; i < nloads; ++i) s += xj[(size_t)li[i] * HC + ch];
  lv[ch] = s;
}

__global__ void lout_k(const float* __restrict__ h2, const float* __restrict__ W,
                       const float* __restrict__ b, float* __restrict__ out, int n)
{
  int tid = blockIdx.x * 256 + threadIdx.x;
  if (tid >= n * 2) return;
  int node = tid >> 1, o = tid & 1;
  const float* hp = h2 + (size_t)node * 64;
  float s = b[o];
#pragma unroll
  for (int k = 0; k < 64; ++k) s += hp[k] * W[k * 2 + o];
  out[tid] = s;
}

// ------------------------------------------------------------ workspace map
static constexpr size_t SZ_NODE = (size_t)NN * HC * 4;        // 51.2 MB
static constexpr size_t SZ_PA   = (size_t)MT * 16 * 512 * 2;  // 51.2 MB (Kmax=512)
static constexpr size_t O_X1   = 0;
static constexpr size_t O_X2   = O_X1 + SZ_NODE;
static constexpr size_t O_X3   = O_X2 + SZ_NODE;
static constexpr size_t O_HST  = O_X3 + SZ_NODE;
static constexpr size_t O_CST  = O_HST + SZ_NODE;
static constexpr size_t O_XJ   = O_CST + SZ_NODE;
static constexpr size_t O_PA   = O_XJ + SZ_NODE;
static constexpr size_t O_HTMP = O_PA + SZ_PA;
static constexpr size_t O_PBUF = O_HTMP + SZ_NODE;            // edge p: 13.6 MB
static constexpr size_t O_AS   = O_PBUF + (size_t)ETOT * 4 * 4;
static constexpr size_t O_AD   = O_AS + (size_t)NN * 16;
static constexpr size_t O_M    = O_AD + (size_t)NN * 16;
static constexpr size_t O_Z    = O_M + (size_t)NN * 16;
static constexpr size_t O_SC   = O_Z + (size_t)NN * 16;       // 3*N f32
static constexpr size_t O_H1   = O_SC + 600064;
static constexpr size_t O_H2   = O_H1 + (size_t)NN * 64 * 4;
static constexpr size_t O_LV   = O_H2 + (size_t)NN * 64 * 4;
static constexpr size_t O_PB0  = O_LV + 1024;
static constexpr size_t O_PB1  = O_PB0 + (size_t)4 * 16 * 512 * 2;
static constexpr size_t O_PB2  = O_PB1 + (size_t)8 * 16 * 512 * 2;
static constexpr size_t O_PBLF = O_PB2 + (size_t)8 * 16 * 512 * 2;
static constexpr size_t O_PBLB = O_PBLF + (size_t)16 * 64 * 512 * 2;
static constexpr size_t O_PBL1 = O_PBLB + (size_t)16 * 64 * 512 * 2;
static constexpr size_t O_PBL2 = O_PBL1 + (size_t)16 * 4 * 512 * 2;
static constexpr size_t O_BF   = O_PBL2 + (size_t)6 * 4 * 512 * 2;
static constexpr size_t O_BB   = O_BF + 4096;
static constexpr size_t WS_NEED = O_BB + 4096;

static inline dim3 nb(size_t n, size_t per) { return dim3((unsigned)((n + per - 1) / per)); }
static inline unsigned gemm_gx(int Mtiles) {
  int pairs = (Mtiles + 1) / 2;
  return (unsigned)((pairs + 3) / 4);
}

extern "C" void kernel_launch(void* const* d_in, const int* in_sizes, int n_in,
                              void* d_out, int out_size, void* d_ws, size_t ws_size,
                              hipStream_t stream)
{
  (void)n_in; (void)out_size;
  if (ws_size < WS_NEED) return;
  char* ws = (char*)d_ws;
  const float* x    = (const float*)d_in[0];
  const int*   ei   = (const int*)d_in[1];
  const int*   li   = (const int*)d_in[2];
  const float* cW[3]  = {(const float*)d_in[3], (const float*)d_in[7],  (const float*)d_in[11]};
  const float* cAs[3] = {(const float*)d_in[4], (const float*)d_in[8],  (const float*)d_in[12]};
  const float* cAd[3] = {(const float*)d_in[5], (const float*)d_in[9],  (const float*)d_in[13]};
  const float* cB[3]  = {(const float*)d_in[6], (const float*)d_in[10], (const float*)d_in[14]};
  const float* Wih_f = (const float*)d_in[15];
  const float* Whh_f = (const float*)d_in[16];
  const float* bih_f = (const float*)d_in[17];
  const float* bhh_f = (const float*)d_in[18];
  const float* Wih_b = (const float*)d_in[19];
  const float* Whh_b = (const float*)d_in[20];
  const float* bih_b = (const float*)d_in[21];
  const float* bhh_b = (const float*)d_in[22];
  const float* attw  = (const float*)d_in[23];
  const float* attb  = (const float*)d_in[24];
  const float* l1W   = (const float*)d_in[25];
  const float* l1b   = (const float*)d_in[26];
  const float* l2W   = (const float*)d_in[27];
  const float* l2b   = (const float*)d_in[28];
  const float* loW   = (const float*)d_in[29];
  const float* lob   = (const float*)d_in[30];
  int nloads = in_sizes[2];

  float* X[3] = {(float*)(ws + O_X1), (float*)(ws + O_X2), (float*)(ws + O_X3)};
  float* HST  = (float*)(ws + O_HST);
  float* CST  = (float*)(ws + O_CST);
  float* XJ   = (float*)(ws + O_XJ);
  unsigned short* PA = (unsigned short*)(ws + O_PA);
  float* HTMP = (float*)(ws + O_HTMP);
  float* PB_E = (float*)(ws + O_PBUF);
  float* AS   = (float*)(ws + O_AS);
  float* AD   = (float*)(ws + O_AD);
  float* Mb   = (float*)(ws + O_M);
  float* Zb   = (float*)(ws + O_Z);
  float* SC   = (float*)(ws + O_SC);
  float* H1   = (float*)(ws + O_H1);
  float* H2   = (float*)(ws + O_H2);
  float* LV   = (float*)(ws + O_LV);
  unsigned short* PBc[3] = {(unsigned short*)(ws + O_PB0), (unsigned short*)(ws + O_PB1),
                            (unsigned short*)(ws + O_PB2)};
  unsigned short* PBLF = (unsigned short*)(ws + O_PBLF);
  unsigned short* PBLB = (unsigned short*)(ws + O_PBLB);
  unsigned short* PBL1 = (unsigned short*)(ws + O_PBL1);
  unsigned short* PBL2 = (unsigned short*)(ws + O_PBL2);
  float* BF = (float*)(ws + O_BF);
  float* BB = (float*)(ws + O_BB);

  // ---- pack all weights to bf16 fragment layout (cheap, once per launch)
  pack_b<<<nb(4 * 16, 8), 256, 0, stream>>>(cW[0], PBc[0], FIN, HC);
  pack_b<<<nb(8 * 16, 8), 256, 0, stream>>>(cW[1], PBc[1], HC, HC);
  pack_b<<<nb(8 * 16, 8), 256, 0, stream>>>(cW[2], PBc[2], HC, HC);
  pack_b_lstm<<<nb(16 * 64, 8), 256, 0, stream>>>(Wih_f, Whh_f, PBLF);
  pack_b_lstm<<<nb(16 * 64, 8), 256, 0, stream>>>(Wih_b, Whh_b, PBLB);
  pack_b<<<nb(16 * 4, 8), 256, 0, stream>>>(l1W, PBL1, 512, 64);
  pack_b<<<nb(6 * 4, 8), 256, 0, stream>>>(l2W, PBL2, 192, 64);
  add_vec_k<<<nb(1024, 256), 256, 0, stream>>>(bih_f, bhh_f, BF, 1024);
  add_vec_k<<<nb(1024, 256), 256, 0, stream>>>(bih_b, bhh_b, BB, 1024);

  // ---- 3 GAT layers
  for (int layer = 0; layer < 3; ++layer) {
    const float* xin = (layer == 0) ? x : X[layer - 1];
    int K = (layer == 0) ? FIN : HC;
    int Ktiles = K / 32;
    pack_a_concat<<<nb((size_t)MT * Ktiles, 8), 256, 0, stream>>>(
        xin, K, nullptr, 0, 0, PA, NN, Ktiles);
    wmma_gemm_bf16<<<dim3(gemm_gx(MT), HC / 64), 128, 0, stream>>>(
        PA, PBc[layer], HTMP, nullptr, MT, Ktiles, HC / 16, HC, 0);
    node_attn_k<<<nb(NN, 8), 256, 0, stream>>>(HTMP, cAs[layer], cAd[layer], AS, AD, NN);
    fill_f32_k<<<nb((size_t)NN * 4, 256), 256, 0, stream>>>(Mb, -3.0e38f, (size_t)NN * 4);
    fill_f32_k<<<nb((size_t)NN * 4, 256), 256, 0, stream>>>(Zb, 0.f, (size_t)NN * 4);
    edge_max_k<<<nb((size_t)ETOT * 4, 256), 256, 0, stream>>>(ei, AS, AD, Mb);
    edge_exp_k<<<nb((size_t)ETOT * 4, 256), 256, 0, stream>>>(ei, AS, AD, Mb, Zb, PB_E);
    bias_rows_k<<<nb((size_t)NN * HC, 256), 256, 0, stream>>>(X[layer], cB[layer],
                                                              (size_t)NN * HC);
    edge_aggr_k<<<dim3(ETOT), 256, 0, stream>>>(ei, HTMP, PB_E, Zb, X[layer]);
    if (layer > 0)
      relu_k<<<nb((size_t)NN * HC, 256), 256, 0, stream>>>(X[layer], (size_t)NN * HC);
  }

  // ---- bidirectional LSTM over [x1,x2,x3], fused gates, streaming JK scores
  fill_ptr_k<<<nb((size_t)3 * NN, 256), 256, 0, stream>>>(SC, attb, (size_t)3 * NN);
  for (int dir = 0; dir < 2; ++dir) {
    const unsigned short* PBl = dir ? PBLB : PBLF;
    const float* bias = dir ? BB : BF;
    const float* aw = attw + dir * HC;
    fill_f32_k<<<nb(SZ_NODE / 4, 256), 256, 0, stream>>>(HST, 0.f, SZ_NODE / 4);
    fill_f32_k<<<nb(SZ_NODE / 4, 256), 256, 0, stream>>>(CST, 0.f, SZ_NODE / 4);
    for (int s = 0; s < 3; ++s) {
      int l = dir ? (2 - s) : s;
      pack_a_concat<<<nb((size_t)MT * 16, 8), 256, 0, stream>>>(
          X[l], HC, HST, HC, 0, PA, NN, 16);
      wmma_lstm_step<<<dim3(gemm_gx(MT), 16), 128, 0, stream>>>(
          PA, PBl, HST, CST, bias, MT);
      score_accum_k<<<nb(NN, 8), 256, 0, stream>>>(HST, aw, SC + (size_t)l * NN, NN);
    }
  }

  // ---- JK softmax combine, load pooling, MLP head
  jk_softmax_k<<<nb((size_t)NN * HC, 256), 256, 0, stream>>>(X[0], X[1], X[2], SC, XJ, NN);
  load_pool_k<<<1, 256, 0, stream>>>(XJ, li, LV, nloads);
  pack_a_concat<<<nb((size_t)MT * 16, 8), 256, 0, stream>>>(XJ, HC, LV, HC, 1, PA, NN, 16);
  wmma_gemm_bf16<<<dim3(gemm_gx(MT), 1), 128, 0, stream>>>(
      PA, PBL1, H1, l1b, MT, 16, 4, 64, 1);
  pack_a_concat<<<nb((size_t)MT * 6, 8), 256, 0, stream>>>(H1, 64, x, FIN, 0, PA, NN, 6);
  wmma_gemm_bf16<<<dim3(gemm_gx(MT), 1), 128, 0, stream>>>(
      PA, PBL2, H2, l2b, MT, 6, 4, 64, 1);
  lout_k<<<nb((size_t)NN * 2, 256), 256, 0, stream>>>(H2, loW, lob, (float*)d_out, NN);
}
